// VaeDecoder_13615046328812
// MI455X (gfx1250) — matrix operand order repaired and checked by reading, not compiled, not run on hardware
//
#include <hip/hip_runtime.h>

// ---------------------------------------------------------------------------
// Problem constants (match reference)
// ---------------------------------------------------------------------------
#define B_    256
#define T_    512
#define OUT_  64
#define COND_ 32
#define LAT_  128
#define HID_  512
#define G4_   2048            // 4*HID
#define KX_   96              // OUT_ + COND_  (x part of LSTM input)
#define KA_   608             // KX_ + HID_    (fused K for the gate GEMM)

typedef __attribute__((ext_vector_type(16))) __bf16 bf16x16;
typedef __attribute__((ext_vector_type(8)))  __bf16 bf16x8;
typedef __attribute__((ext_vector_type(8)))  float  f32x8;

union FragBF { bf16x16 v; bf16x8 h[2]; };

// B operand (K x 16), 16-bit elems: lane<16 holds N=lane, K=k0..k0+7, k0+16..k0+23;
// lane>=16 holds N=lane-16, K=k0+8..k0+15, k0+24..k0+31; K packed 2/VGPR in order.
// Source row-major W[N][K] -> each lane reads two runs of 8 contiguous bf16, 16 apart.
__device__ __forceinline__ bf16x16 load_b_frag(const __bf16* __restrict__ W,
                                               int strideElems, int n0, int k0,
                                               int lane) {
  const __bf16* p = W + (size_t)(n0 + (lane & 15)) * strideElems
                      + (k0 + ((lane >> 4) << 3));
  FragBF u;
  u.h[0] = *reinterpret_cast<const bf16x8*>(p);
  u.h[1] = *reinterpret_cast<const bf16x8*>(p + 16);
  return u.v;
}

// A operand (16 x 32), 16-bit elems, row-major panel in LDS:
// lane<16: M=lane, VGPR0-3: K=k0..k0+7, VGPR4-7: K=k0+16..k0+23
// lane>=16: M=lane-16, VGPR0-3: K=k0+8..k0+15, VGPR4-7: K=k0+24..k0+31
__device__ __forceinline__ bf16x16 load_a_frag(const __bf16* A, int strideElems,
                                               int k0, int lane) {
  const int m   = lane & 15;
  const int sel = (lane >> 4) << 3;   // 0 or 8 element offset
  const __bf16* row = A + m * strideElems + k0;
  FragBF u;
  u.h[0] = *reinterpret_cast<const bf16x8*>(row + sel);
  u.h[1] = *reinterpret_cast<const bf16x8*>(row + 16 + sel);
  return u.v;
}

__device__ __forceinline__ float fsig(float x) {
  return 1.0f / (1.0f + __expf(-x));
}
__device__ __forceinline__ float fast_tanh(float x) {
  float e = __expf(-2.0f * fabsf(x));          // (1-e)/(1+e) in [0,1)
  float t = (1.0f - e) / (1.0f + e);
  return copysignf(t, x);
}

// ---------------------------------------------------------------------------
// Prep: fuse + convert weights to bf16, pre-sum biases (runs once per launch)
// ---------------------------------------------------------------------------
__global__ void vae_prep_kernel(const float* __restrict__ Wih,
                                const float* __restrict__ Whh,
                                const float* __restrict__ Wlh_f,
                                const float* __restrict__ Wlc_f,
                                const float* __restrict__ Wout_f,
                                const float* __restrict__ bih,
                                const float* __restrict__ bhh,
                                __bf16* __restrict__ wg,
                                __bf16* __restrict__ wlh,
                                __bf16* __restrict__ wlc,
                                __bf16* __restrict__ wout,
                                float* __restrict__ bsum) {
  const int stride = blockDim.x * gridDim.x;
  const int tid = blockIdx.x * blockDim.x + threadIdx.x;
  for (int i = tid; i < G4_ * KA_; i += stride) {
    int n = i / KA_, k = i - n * KA_;
    float v = (k < KX_) ? Wih[n * KX_ + k] : Whh[n * HID_ + (k - KX_)];
    wg[i] = (__bf16)v;
  }
  for (int i = tid; i < HID_ * LAT_; i += stride) {
    wlh[i] = (__bf16)Wlh_f[i];
    wlc[i] = (__bf16)Wlc_f[i];
  }
  for (int i = tid; i < OUT_ * HID_; i += stride) wout[i] = (__bf16)Wout_f[i];
  for (int i = tid; i < G4_; i += stride) bsum[i] = bih[i] + bhh[i];
}

// ---------------------------------------------------------------------------
// Main scan kernel: 16 workgroups (one 16-row batch tile each), 1024 threads.
// Wave w (of 32) owns hidden units [16w, 16w+16): its i/f/g/o accumulator
// tiles and the persistent fp32 cell state c live entirely in VGPRs.
// LDS holds a double-buffered 16x608 bf16 [x | h] panel; one barrier / step.
// ---------------------------------------------------------------------------
__global__ __launch_bounds__(1024, 1) void vae_lstm_scan(
    const float* __restrict__ latent, const float* __restrict__ cond,
    const float* __restrict__ target, const float* __restrict__ b_lh,
    const float* __restrict__ b_lc, const float* __restrict__ b_out,
    const __bf16* __restrict__ Wg, const __bf16* __restrict__ Wlh,
    const __bf16* __restrict__ Wlc, const __bf16* __restrict__ Wout,
    const float* __restrict__ bsum, float* __restrict__ out) {
  __shared__ __align__(16) __bf16 sA[2][16 * KA_];   // [x(96) | h(512)] panels
  __shared__ __align__(16) __bf16 sLat[16 * LAT_];   // latent panel (prologue)

  const int tid  = threadIdx.x;
  const int wave = tid >> 5;
  const int lane = tid & 31;
  const int b0   = blockIdx.x * 16;         // batch tile base
  const int n0   = wave * 16;               // hidden-unit base for this wave
  const int nlo  = lane & 15;
  const int mof  = (lane >> 4) << 3;        // C/D layout: M = r + mof

  // --- prologue: out[:,0,:]=1, x0 = [ones | cond], latent -> LDS bf16 ------
  {
    int m = tid >> 6, j = tid & 63;
    out[((size_t)(b0 + m) * T_ + 0) * OUT_ + j] = 1.0f;   // SOH row
    sA[0][m * KA_ + j] = (__bf16)1.0f;                    // x0 prev = ones
  }
  if (tid < 512) {                                        // cond, both buffers
    int m = tid >> 5, j = tid & 31;
    __bf16 cv = (__bf16)cond[(b0 + m) * COND_ + j];
    sA[0][m * KA_ + OUT_ + j] = cv;
    sA[1][m * KA_ + OUT_ + j] = cv;
  }
  for (int i = tid; i < 16 * LAT_; i += 1024)
    sLat[i] = (__bf16)latent[(size_t)b0 * LAT_ + i];
  __syncthreads();

  // --- h0 / c0 via bf16 WMMA: (16x128) @ (128x16) per wave ----------------
  f32x8 creg;
  {
    const float blh = b_lh[n0 + nlo];
    const float blc = b_lc[n0 + nlo];
    f32x8 ah, ac;
#pragma unroll
    for (int e = 0; e < 8; ++e) { ah[e] = blh; ac[e] = blc; }
#pragma unroll
    for (int k0 = 0; k0 < LAT_; k0 += 32) {
      bf16x16 a  = load_a_frag(sLat, LAT_, k0, lane);
      bf16x16 bh = load_b_frag(Wlh, LAT_, n0, k0, lane);
      bf16x16 bc = load_b_frag(Wlc, LAT_, n0, k0, lane);
      ah = __builtin_amdgcn_wmma_f32_16x16x32_bf16(false, a, false, bh,
                                                   (short)0, ah, false, false);
      ac = __builtin_amdgcn_wmma_f32_16x16x32_bf16(false, a, false, bc,
                                                   (short)0, ac, false, false);
    }
    creg = ac;
#pragma unroll
    for (int r = 0; r < 8; ++r)                           // h0 -> panel 0
      sA[0][(r + mof) * KA_ + KX_ + n0 + nlo] = (__bf16)ah[r];
  }
  __syncthreads();

  // Per-wave gate biases (b_ih + b_hh), output bias (waves 0..3)
  const float bg0 = bsum[0 * HID_ + n0 + nlo];
  const float bg1 = bsum[1 * HID_ + n0 + nlo];
  const float bg2 = bsum[2 * HID_ + n0 + nlo];
  const float bg3 = bsum[3 * HID_ + n0 + nlo];
  const float bo  = (wave < 4) ? b_out[wave * 16 + nlo] : 0.0f;

  // --- 511 recurrent steps ------------------------------------------------
  for (int t = 0; t < T_ - 1; ++t) {
    const __bf16* rbuf = sA[t & 1];
    __bf16* wbuf = sA[(t & 1) ^ 1];

    f32x8 acc[4];
#pragma unroll
    for (int e = 0; e < 8; ++e) {
      acc[0][e] = bg0; acc[1][e] = bg1; acc[2][e] = bg2; acc[3][e] = bg3;
    }
    // gates = [x|h] @ [W_ih|W_hh].T   (K = 608, 19 WMMA k-steps x 4 gates)
    for (int k0 = 0; k0 < KA_; k0 += 32) {
      bf16x16 a = load_a_frag(rbuf, KA_, k0, lane);
#pragma unroll
      for (int g = 0; g < 4; ++g) {
        bf16x16 b = load_b_frag(Wg, KA_, g * HID_ + n0, k0, lane);
        acc[g] = __builtin_amdgcn_wmma_f32_16x16x32_bf16(
            false, a, false, b, (short)0, acc[g], false, false);
      }
    }
    // LSTM cell math in fp32 registers; h -> next panel (bf16)
#pragma unroll
    for (int r = 0; r < 8; ++r) {
      float ig = fsig(acc[0][r]);
      float fg = fsig(acc[1][r]);
      float gg = fast_tanh(acc[2][r]);
      float og = fsig(acc[3][r]);
      float cn = fg * creg[r] + ig * gg;
      creg[r]  = cn;
      float hv = og * fast_tanh(cn);
      wbuf[(r + mof) * KA_ + KX_ + n0 + nlo] = (__bf16)hv;
    }
    // teacher-forcing prefetch: xs[t+1] = target[:, t, :]
    {
      int m = tid >> 6, j = tid & 63;
      wbuf[m * KA_ + j] =
          (__bf16)target[((size_t)(b0 + m) * T_ + t) * OUT_ + j];
    }
    __syncthreads();

    // output projection of h_t (waves 0..3): (16x512) @ (512x16) each
    if (wave < 4) {
      f32x8 oacc;
#pragma unroll
      for (int e = 0; e < 8; ++e) oacc[e] = bo;
      const __bf16* hb = wbuf + KX_;
#pragma unroll 4
      for (int k0 = 0; k0 < HID_; k0 += 32) {
        bf16x16 a = load_a_frag(hb, KA_, k0, lane);
        bf16x16 b = load_b_frag(Wout, HID_, wave * 16, k0, lane);
        oacc = __builtin_amdgcn_wmma_f32_16x16x32_bf16(
            false, a, false, b, (short)0, oacc, false, false);
      }
#pragma unroll
      for (int r = 0; r < 8; ++r)
        out[((size_t)(b0 + r + mof) * T_ + (t + 1)) * OUT_ + wave * 16 + nlo] =
            oacc[r];
    }
  }
}

// ---------------------------------------------------------------------------
extern "C" void kernel_launch(void* const* d_in, const int* in_sizes, int n_in,
                              void* d_out, int out_size, void* d_ws,
                              size_t ws_size, hipStream_t stream) {
  const float* latent = (const float*)d_in[0];
  const float* cond   = (const float*)d_in[1];
  const float* target = (const float*)d_in[2];
  const float* Wlh_f  = (const float*)d_in[3];
  const float* b_lh   = (const float*)d_in[4];
  const float* Wlc_f  = (const float*)d_in[5];
  const float* b_lc   = (const float*)d_in[6];
  const float* Wih    = (const float*)d_in[7];
  const float* Whh    = (const float*)d_in[8];
  const float* bih    = (const float*)d_in[9];
  const float* bhh    = (const float*)d_in[10];
  const float* Wout_f = (const float*)d_in[11];
  const float* b_out  = (const float*)d_in[12];
  (void)in_sizes; (void)n_in; (void)out_size; (void)ws_size;

  char* ws = (char*)d_ws;
  size_t off = 0;
  auto carve = [&](size_t bytes) {
    void* p = ws + off;
    off = (off + bytes + 255) & ~(size_t)255;
    return p;
  };
  __bf16* wg   = (__bf16*)carve((size_t)G4_ * KA_ * 2);    // ~2.49 MB
  __bf16* wlh  = (__bf16*)carve((size_t)HID_ * LAT_ * 2);  // 128 KB
  __bf16* wlc  = (__bf16*)carve((size_t)HID_ * LAT_ * 2);  // 128 KB
  __bf16* wout = (__bf16*)carve((size_t)OUT_ * HID_ * 2);  // 64 KB
  float*  bsum = (float*)carve((size_t)G4_ * 4);           // 8 KB

  vae_prep_kernel<<<512, 256, 0, stream>>>(Wih, Whh, Wlh_f, Wlc_f, Wout_f,
                                           bih, bhh, wg, wlh, wlc, wout, bsum);
  vae_lstm_scan<<<B_ / 16, 1024, 0, stream>>>(
      latent, cond, target, b_lh, b_lc, b_out, wg, wlh, wlc, wout, bsum,
      (float*)d_out);
}
